// GroupQueryAttention_52536039965331
// MI455X (gfx1250) — matrix operand order repaired and checked by reading, not compiled, not run on hardware
//
#include <hip/hip_runtime.h>

#define BATCH  2
#define SLEN   2048
#define HDIM   2048
#define NHEADS 16
#define GROUPS 4
#define HEADD  128
#define QREP   4
#define KVDIM  512

typedef __attribute__((ext_vector_type(16))) __bf16 v16bf;
typedef __attribute__((ext_vector_type(8)))  __bf16 v8bf;
typedef __attribute__((ext_vector_type(8)))  float  v8f;
typedef __attribute__((ext_vector_type(4)))  unsigned int u32x4;
typedef __attribute__((ext_vector_type(4)))  int i32x4;
typedef __attribute__((ext_vector_type(8)))  int i32x8;

// Toolchain-dependent arity of the TDM builtin (see CDNA5_HIP.md):
//   ROCm 7.2 / clang-22: 5 args; amdgpu-toolchain / clang-23: 6 args.
#if __has_include(<hip/amd_detail/amd_gfx1250_TDM.h>)
#define TDM_LOAD(g0, g1) \
  __builtin_amdgcn_tensor_load_to_lds((g0), (g1), (i32x4)0, (i32x4)0, (i32x8)0, 0)
#else
#define TDM_LOAD(g0, g1) \
  __builtin_amdgcn_tensor_load_to_lds((g0), (g1), (i32x4)0, (i32x4)0, 0)
#endif

// ---------------------------------------------------------------------------
// Fragment loaders for V_WMMA_F32_16X16X32_BF16 (wave32).
// A (16xK=32, row-major source):
//   lane L: m = L&15, h = L>>4; elems 0..15 = A[m][k0+16h .. k0+16h+15]
// B (K=32 x 16) from an N-major ("transposed") source BT[n][k]:
//   lane L: n = n0 + (L&15), h = L>>4; elems 0..15 = B[k0+16h+e][n] = BT[n][k0+16h+e]
// Both are two contiguous 16-byte loads per lane.
// ---------------------------------------------------------------------------
static __device__ inline v16bf load_a_frag(const __bf16* __restrict__ A, int lda,
                                           int m0, int k0, int lane) {
  int m = m0 + (lane & 15);
  int h = (lane >> 4) & 1;
  const __bf16* p = A + (size_t)m * lda + k0;
  union { v16bf v; v8bf h2[2]; } u;
  u.h2[0] = *(const v8bf*)(p + 16 * h);
  u.h2[1] = *(const v8bf*)(p + 16 * h + 8);
  return u.v;
}

static __device__ inline v16bf load_bt_frag(const __bf16* __restrict__ BT, int ldb,
                                            int n0, int k0, int lane) {
  int n = n0 + (lane & 15);
  int h = (lane >> 4) & 1;
  const __bf16* p = BT + (size_t)n * ldb + k0 + 16 * h;
  union { v16bf v; v8bf h2[2]; } u;
  u.h2[0] = *(const v8bf*)(p);
  u.h2[1] = *(const v8bf*)(p + 8);
  return u.v;
}

// ---------------------------------------------------------------------------
// f32 -> bf16, 8 elements per thread (2x b128 in, b128 out)
// ---------------------------------------------------------------------------
__global__ __launch_bounds__(256) void cvt_bf16(const float* __restrict__ src,
                                                __bf16* __restrict__ dst, int n) {
  int i = (blockIdx.x * blockDim.x + threadIdx.x) * 8;
  if (i + 8 <= n) {
    v8f x = *(const v8f*)(src + i);
    v8bf y;
#pragma unroll
    for (int e = 0; e < 8; ++e) y[e] = (__bf16)x[e];
    *(v8bf*)(dst + i) = y;
  }
}

// W[k][n] (f32, K x N row-major) -> WT[n][k] (bf16, N-major).
// LDS-tiled 32x32 transpose: coalesced reads AND writes.
__global__ __launch_bounds__(256) void transpose_bf16(const float* __restrict__ W,
                                                      __bf16* __restrict__ WT,
                                                      int K, int N) {
  __shared__ float tile[32][33];
  int n0 = blockIdx.x * 32;
  int k0 = blockIdx.y * 32;
  int tx = threadIdx.x, ty = threadIdx.y; // 32 x 8
#pragma unroll
  for (int j = 0; j < 4; ++j) {
    int k = k0 + ty + j * 8;
    tile[ty + j * 8][tx] = W[(size_t)k * N + n0 + tx];
  }
  __syncthreads();
#pragma unroll
  for (int j = 0; j < 4; ++j) {
    int n = n0 + ty + j * 8;
    WT[(size_t)n * K + k0 + tx] = (__bf16)tile[tx][ty + j * 8];
  }
}

// ---------------------------------------------------------------------------
// Register-blocked WMMA GEMM:  C[M,N] = A[M,K] * B[K,N] + bias[N]
// Each wave computes a 64x64 block: 4 A-frags x 4 B-frags -> 16 WMMAs per
// 32-wide k-step (1 b128 load per WMMA).  Block = 4 waves covering 64(M)x256(N).
// ---------------------------------------------------------------------------
template <int MODE>
__global__ __launch_bounds__(128) void gqa_gemm(const __bf16* __restrict__ A,
                                                const __bf16* __restrict__ BT,
                                                const float* __restrict__ bias,
                                                void* __restrict__ dst,
                                                int M, int N, int Kd) {
  int lane = threadIdx.x & 31;
  int wave = threadIdx.x >> 5;
  int m0 = blockIdx.x * 64;
  int n0 = (blockIdx.y * 4 + wave) * 64;

  v8f acc[4][4];
#pragma unroll
  for (int mt = 0; mt < 4; ++mt)
#pragma unroll
    for (int nt = 0; nt < 4; ++nt) acc[mt][nt] = (v8f){};

  for (int k = 0; k < Kd; k += 32) {
    if (k + 64 < Kd) { // prefetch two k-iterations ahead (global_prefetch_b8)
      __builtin_prefetch(A + (size_t)(m0 + (lane & 15)) * Kd + k + 64, 0, 0);
      __builtin_prefetch(BT + (size_t)(n0 + (lane & 15)) * Kd + k + 64, 0, 0);
    }
    v16bf a[4];
#pragma unroll
    for (int mt = 0; mt < 4; ++mt)
      a[mt] = load_a_frag(A, Kd, m0 + mt * 16, k, lane);
#pragma unroll
    for (int nt = 0; nt < 4; ++nt) {
      v16bf b = load_bt_frag(BT, Kd, n0 + nt * 16, k, lane);
#pragma unroll
      for (int mt = 0; mt < 4; ++mt)
        acc[mt][nt] = __builtin_amdgcn_wmma_f32_16x16x32_bf16(
            false, a[mt], false, b, (short)0, acc[mt][nt], false, false);
    }
  }

  int half = lane >> 4;
#pragma unroll
  for (int nt = 0; nt < 4; ++nt) {
    int nc = n0 + nt * 16 + (lane & 15);
    float bi = bias[nc];
#pragma unroll
    for (int mt = 0; mt < 4; ++mt) {
#pragma unroll
      for (int i = 0; i < 8; ++i) {
        int m = m0 + mt * 16 + i + 8 * half;
        float val = acc[mt][nt][i] + bi;
        int bb = m / SLEN, s = m % SLEN;
        if (MODE == 0) {
          int head = nc / HEADD, d = nc % HEADD;
          ((__bf16*)dst)[(((size_t)bb * NHEADS + head) * SLEN + s) * HEADD + d] = (__bf16)val;
        } else if (MODE == 1) {
          int g = nc / HEADD, d = nc % HEADD;
          ((__bf16*)dst)[(((size_t)bb * GROUPS + g) * SLEN + s) * HEADD + d] = (__bf16)val;
        } else if (MODE == 2) {
          int g = nc / HEADD, d = nc % HEADD;
          ((__bf16*)dst)[(((size_t)bb * GROUPS + g) * HEADD + d) * SLEN + s] = (__bf16)val;
        } else {
          ((float*)dst)[(size_t)m * N + nc] = val;
        }
      }
    }
  }
}

// ---------------------------------------------------------------------------
// TDM issue: stage one 32-key chunk of K (32x128, row-major) and V^T
// (128x32, key-contiguous) into LDS.  D# built per CDNA5 ISA sec. 8.3/8.4:
//   group0: count=1 | lds_addr | global_addr[31:0] | global_addr[56:32] + type=2
//   group1: data_size=1(2B) | tensor_dim0/1 | tile_dim0/1 | tensor_dim0_stride
// 2-D tile -> descriptor groups 2/3 zero.
// ---------------------------------------------------------------------------
static __device__ inline void tdm_stage_kv(const __bf16* kg, const __bf16* vg,
                                           unsigned klds, unsigned vlds) {
  u32x4 g0;
  i32x8 g1;
  // ---- K tile: tile_dim0 = 128 (d, contiguous), tile_dim1 = 32 (keys)
  unsigned long long ka = (unsigned long long)(uintptr_t)kg;
  g0[0] = 1u;                                   // count = 1
  g0[1] = klds;                                 // lds_addr
  g0[2] = (unsigned)ka;                         // global_addr[31:0]
  g0[3] = (unsigned)((ka >> 32) & 0x1FFFFFFu) | (2u << 30); // addr hi | type=2
  g1[0] = (int)(1u << 16);                      // data_size = 1 (2 bytes)
  g1[1] = (int)((unsigned)HEADD << 16);         // tensor_dim0 = 128
  g1[2] = (int)((unsigned)SLEN << 16);          // tensor_dim1 = 2048
  g1[3] = (int)((unsigned)HEADD << 16);         // tile_dim0 = 128
  g1[4] = 32;                                   // tile_dim1 = 32
  g1[5] = HEADD;                                // tensor_dim0_stride = 128
  g1[6] = 0;
  g1[7] = 0;
  TDM_LOAD(g0, g1);
  // ---- V tile: tile_dim0 = 32 (keys, contiguous), tile_dim1 = 128 (d rows)
  unsigned long long va = (unsigned long long)(uintptr_t)vg;
  g0[1] = vlds;
  g0[2] = (unsigned)va;
  g0[3] = (unsigned)((va >> 32) & 0x1FFFFFFu) | (2u << 30);
  g1[1] = (int)((unsigned)SLEN << 16);          // tensor_dim0 = 2048
  g1[2] = (int)((unsigned)HEADD << 16);         // tensor_dim1 = 128
  g1[3] = (int)(32u << 16);                     // tile_dim0 = 32
  g1[4] = HEADD;                                // tile_dim1 = 128
  g1[5] = SLEN;                                 // tensor_dim0_stride = 2048
  TDM_LOAD(g0, g1);
}

// ---------------------------------------------------------------------------
// Flash attention, one wave per (b, head, 16-query tile).
// K/V chunks are staged into LDS by the Tensor Data Mover, double-buffered:
// chunk c+1 is in flight (TENSORcnt) while WMMAs consume chunk c from LDS.
// ---------------------------------------------------------------------------
__global__ __launch_bounds__(32) void gqa_attn(const __bf16* __restrict__ Q,
                                               const __bf16* __restrict__ Kb,
                                               const __bf16* __restrict__ Vt,
                                               __bf16* __restrict__ AO) {
  const float scale = 0.08838834764831845f; // 1/sqrt(128)
  int lane = threadIdx.x & 31;
  int q0 = blockIdx.x * 16;
  int bh = (int)blockIdx.y;
  int b = bh / NHEADS, h = bh % NHEADS, g = h / QREP;

  const __bf16* Qh = Q  + (((size_t)b * NHEADS + h) * SLEN) * HEADD;
  const __bf16* Kh = Kb + (((size_t)b * GROUPS + g) * SLEN) * HEADD;
  const __bf16* Vh = Vt + (((size_t)b * GROUPS + g) * HEADD) * SLEN;

  int half = lane >> 4;
  int nc = lane & 15;
  int rb = 8 * half; // rows held by this lane: rb+0 .. rb+7

  __shared__ __bf16 kbuf[2][32 * HEADD];  // 2 x 8 KB, [key][d]
  __shared__ __bf16 vbuf[2][HEADD * 32];  // 2 x 8 KB, [d][key]
  __shared__ __bf16 pl[16 * 32];

  v16bf qf[4];
#pragma unroll
  for (int kk = 0; kk < 4; ++kk) qf[kk] = load_a_frag(Qh, HEADD, q0, kk * 32, lane);

  float m_run[8], l_run[8];
  v8f O[8];
#pragma unroll
  for (int i = 0; i < 8; ++i) { m_run[i] = -3.0e38f; l_run[i] = 0.0f; }
#pragma unroll
  for (int t = 0; t < 8; ++t) O[t] = (v8f){};

  int nchunks = (q0 + 16 + 31) / 32;

  // prologue: stage chunk 0
  tdm_stage_kv(Kh, Vh,
               (unsigned)(uintptr_t)&kbuf[0][0],
               (unsigned)(uintptr_t)&vbuf[0][0]);

  for (int c = 0; c < nchunks; ++c) {
    int kb = c * 32;
    int cur = c & 1;
    if (c + 1 < nchunks) { // stage next chunk into the other buffer, then
      tdm_stage_kv(Kh + (size_t)(kb + 32) * HEADD, Vh + (kb + 32),
                   (unsigned)(uintptr_t)&kbuf[cur ^ 1][0],
                   (unsigned)(uintptr_t)&vbuf[cur ^ 1][0]);
      __builtin_amdgcn_s_wait_tensorcnt(2); // wait only for chunk c (2 ops)
    } else {
      __builtin_amdgcn_s_wait_tensorcnt(0);
    }
    const __bf16* kc = &kbuf[cur][0];
    const __bf16* vc = &vbuf[cur][0];

    // scores for two 16-key tiles (K fragments now from LDS)
    v8f s0 = {}, s1 = {};
#pragma unroll
    for (int kk = 0; kk < 4; ++kk) {
      v16bf kf = load_bt_frag(kc, HEADD, 0, kk * 32, lane);
      s0 = __builtin_amdgcn_wmma_f32_16x16x32_bf16(false, qf[kk], false, kf,
                                                   (short)0, s0, false, false);
    }
#pragma unroll
    for (int kk = 0; kk < 4; ++kk) {
      v16bf kf = load_bt_frag(kc, HEADD, 16, kk * 32, lane);
      s1 = __builtin_amdgcn_wmma_f32_16x16x32_bf16(false, qf[kk], false, kf,
                                                   (short)0, s1, false, false);
    }

    // scale + causal mask + per-row max (rows rb..rb+7, cols across 16 lanes)
    float p0[8], p1[8], mx[8], ps[8], f[8];
#pragma unroll
    for (int i = 0; i < 8; ++i) {
      int qrow = q0 + rb + i;
      float a = s0[i] * scale; if (kb + nc      > qrow) a = -3.0e38f;
      float c2 = s1[i] * scale; if (kb + 16 + nc > qrow) c2 = -3.0e38f;
      p0[i] = a; p1[i] = c2; mx[i] = fmaxf(a, c2);
    }
#pragma unroll
    for (int msk = 1; msk < 16; msk <<= 1) {
#pragma unroll
      for (int i = 0; i < 8; ++i) mx[i] = fmaxf(mx[i], __shfl_xor(mx[i], msk, 32));
    }
#pragma unroll
    for (int i = 0; i < 8; ++i) {
      float mnew = fmaxf(m_run[i], mx[i]);
      f[i] = __expf(m_run[i] - mnew);
      m_run[i] = mnew;
      p0[i] = __expf(p0[i] - mnew);
      p1[i] = __expf(p1[i] - mnew);
      ps[i] = p0[i] + p1[i];
    }
#pragma unroll
    for (int msk = 1; msk < 16; msk <<= 1) {
#pragma unroll
      for (int i = 0; i < 8; ++i) ps[i] += __shfl_xor(ps[i], msk, 32);
    }
#pragma unroll
    for (int i = 0; i < 8; ++i) l_run[i] = l_run[i] * f[i] + ps[i];
#pragma unroll
    for (int t = 0; t < 8; ++t) {
#pragma unroll
      for (int i = 0; i < 8; ++i) O[t][i] *= f[i];
    }

    // stage P (C/D layout) -> LDS -> A-fragment layout
#pragma unroll
    for (int i = 0; i < 8; ++i) {
      pl[(rb + i) * 32 + nc]      = (__bf16)p0[i];
      pl[(rb + i) * 32 + 16 + nc] = (__bf16)p1[i];
    }
    __syncthreads();
    v16bf pf = load_a_frag(pl, 32, 0, 0, lane);
    __syncthreads();

    // O += P @ V  (8 d-tiles of 16; V fragments from LDS)
#pragma unroll
    for (int t = 0; t < 8; ++t) {
      int d = t * 16 + nc;
      const __bf16* p = vc + d * 32 + 16 * half;
      union { v16bf v; v8bf h2[2]; } u;
      u.h2[0] = *(const v8bf*)(p);
      u.h2[1] = *(const v8bf*)(p + 8);
      O[t] = __builtin_amdgcn_wmma_f32_16x16x32_bf16(false, pf, false, u.v,
                                                     (short)0, O[t], false, false);
    }
  }

  // normalize and store to AO[B*S][H] at column h*HD + d
#pragma unroll
  for (int t = 0; t < 8; ++t) {
#pragma unroll
    for (int i = 0; i < 8; ++i) {
      float val = O[t][i] / l_run[i];
      int row = q0 + rb + i;
      AO[((size_t)b * SLEN + row) * HDIM + h * HEADD + t * 16 + nc] = (__bf16)val;
    }
  }
}

// ---------------------------------------------------------------------------
extern "C" void kernel_launch(void* const* d_in, const int* in_sizes, int n_in,
                              void* d_out, int out_size, void* d_ws, size_t ws_size,
                              hipStream_t stream) {
  const float* X  = (const float*)d_in[0];
  const float* Wq = (const float*)d_in[1];
  const float* bq = (const float*)d_in[2];
  const float* Wk = (const float*)d_in[3];
  const float* bk = (const float*)d_in[4];
  const float* Wv = (const float*)d_in[5];
  const float* bv = (const float*)d_in[6];
  const float* Wo = (const float*)d_in[7];
  const float* bo = (const float*)d_in[8];
  // d_in[9] = causal mask (implemented analytically in-kernel)

  char* ws = (char*)d_ws;
  size_t off = 0;
  __bf16* Xbf = (__bf16*)(ws + off); off += (size_t)BATCH * SLEN * HDIM * 2;
  __bf16* WqT = (__bf16*)(ws + off); off += (size_t)HDIM * HDIM * 2;
  __bf16* WkT = (__bf16*)(ws + off); off += (size_t)KVDIM * HDIM * 2;
  __bf16* WvT = (__bf16*)(ws + off); off += (size_t)KVDIM * HDIM * 2;
  __bf16* WoT = (__bf16*)(ws + off); off += (size_t)HDIM * HDIM * 2;
  __bf16* Qbf = (__bf16*)(ws + off); off += (size_t)BATCH * NHEADS * SLEN * HEADD * 2;
  __bf16* Kbf = (__bf16*)(ws + off); off += (size_t)BATCH * GROUPS * SLEN * HEADD * 2;
  __bf16* Vtb = (__bf16*)(ws + off); off += (size_t)BATCH * GROUPS * HEADD * SLEN * 2;
  __bf16* AO  = (__bf16*)(ws + off); off += (size_t)BATCH * SLEN * HDIM * 2;

  const int M = BATCH * SLEN; // 4096 rows

  // 1) inputs -> bf16 (weights transposed to N-major, LDS-tiled)
  {
    int n = BATCH * SLEN * HDIM;
    cvt_bf16<<<(n / 8 + 255) / 256, 256, 0, stream>>>(X, Xbf, n);
    dim3 tb(32, 8);
    transpose_bf16<<<dim3(HDIM / 32,  HDIM / 32), tb, 0, stream>>>(Wq, WqT, HDIM, HDIM);
    transpose_bf16<<<dim3(KVDIM / 32, HDIM / 32), tb, 0, stream>>>(Wk, WkT, HDIM, KVDIM);
    transpose_bf16<<<dim3(KVDIM / 32, HDIM / 32), tb, 0, stream>>>(Wv, WvT, HDIM, KVDIM);
    transpose_bf16<<<dim3(HDIM / 32,  HDIM / 32), tb, 0, stream>>>(Wo, WoT, HDIM, HDIM);
  }

  // 2) projections (WMMA bf16, f32 accumulate); wave tile 64x64, block 64x256
  gqa_gemm<0><<<dim3(M / 64, HDIM / 256),  128, 0, stream>>>(Xbf, WqT, bq, Qbf, M, HDIM,  HDIM);
  gqa_gemm<1><<<dim3(M / 64, KVDIM / 256), 128, 0, stream>>>(Xbf, WkT, bk, Kbf, M, KVDIM, HDIM);
  gqa_gemm<2><<<dim3(M / 64, KVDIM / 256), 128, 0, stream>>>(Xbf, WvT, bv, Vtb, M, KVDIM, HDIM);

  // 3) causal flash attention (WMMA + TDM double-buffered K/V staging)
  gqa_attn<<<dim3(SLEN / 16, BATCH * NHEADS), 32, 0, stream>>>(Qbf, Kbf, Vtb, AO);

  // 4) output projection -> f32 d_out
  gqa_gemm<3><<<dim3(M / 64, HDIM / 256), 128, 0, stream>>>(AO, WoT, bo, d_out, M, HDIM, HDIM);
}
